// MicroSegHead_13709535609046
// MI455X (gfx1250) — matrix operand order repaired and checked by reading, not compiled, not run on hardware
//
#include <hip/hip_runtime.h>

// ---------------------------------------------------------------------------
// MicroSegHead for MI455X (gfx1250, wave32).
// 9x [4096 x 2048 x 2048] GEMM+BN+ReLU fused, bf16 WMMA with f32 accumulate
// (v_wmma_f32_16x16x32_bf16), double-buffered LDS (padded rows -> no bank
// conflicts). A tile (already bf16) streamed global->LDS with CDNA5 async
// copies (ASYNCcnt); W tile converted f32->bf16 in flight using native cvt
// (packed (__bf16) casts). Tiny classifier GEMMs done with a vectorized dot
// kernel (<<1% of FLOPs).
// ---------------------------------------------------------------------------

#define M_TOT 4096
#define C_TOT 2048
#define TILE_M 128
#define TILE_N 128
#define TILE_K 32
#define LDS_STRIDE 40   // 32 + 8 pad (ushorts): 80B row stride -> conflict-free ds_load_b128
#define NK_STEPS (C_TOT / TILE_K)

#if defined(__has_builtin)
#if __has_builtin(__builtin_amdgcn_global_load_async_to_lds_b128)
#define USE_ASYNC_COPY 1
#else
#define USE_ASYNC_COPY 0
#endif
#else
#define USE_ASYNC_COPY 0
#endif

typedef __attribute__((ext_vector_type(16))) __bf16 v16bf;
typedef __attribute__((ext_vector_type(8)))  __bf16 bf16x8;
typedef __attribute__((ext_vector_type(2)))  __bf16 bf16x2;
typedef __attribute__((ext_vector_type(8)))  float  v8f;
typedef __attribute__((ext_vector_type(4)))  int    v4i;

#if USE_ASYNC_COPY
typedef __attribute__((address_space(1))) v4i glob_v4i;
typedef __attribute__((address_space(3))) v4i lds_v4i;
#endif

// Native conversions: let ISel pick v_cvt_pk_bf16_f32-class ops instead of
// the bit-twiddling sequence (which scalarized into v_mov_b16 storms).
__device__ __forceinline__ unsigned pack_bf16x2(float lo, float hi) {
  bf16x2 t;
  t.x = (__bf16)lo;
  t.y = (__bf16)hi;
  return __builtin_bit_cast(unsigned, t);
}
__device__ __forceinline__ unsigned short f32_to_bf16(float f) {
  __bf16 h = (__bf16)f;
  return __builtin_bit_cast(unsigned short, h);
}
__device__ __forceinline__ float bf16_to_f32(unsigned short h) {
  unsigned u = ((unsigned)h) << 16;
  return __builtin_bit_cast(float, u);
}

__device__ __forceinline__ void wait_async_zero() {
#if USE_ASYNC_COPY
#if __has_builtin(__builtin_amdgcn_s_wait_asynccnt)
  __builtin_amdgcn_s_wait_asynccnt(0);
#else
  asm volatile("s_wait_asynccnt 0" ::: "memory");
#endif
#endif
}

// ---------------------------------------------------------------------------
// features f32 -> bf16 (one-time, 16.8 MB)
// ---------------------------------------------------------------------------
__global__ void cvt_f32_to_bf16(const float4* __restrict__ src,
                                uint2* __restrict__ dst, int n4) {
  int i = blockIdx.x * blockDim.x + threadIdx.x;
  if (i >= n4) return;
  float4 f = src[i];
  uint2 o;
  o.x = pack_bf16x2(f.x, f.y);
  o.y = pack_bf16x2(f.z, f.w);
  dst[i] = o;
}

// ---------------------------------------------------------------------------
// D[m,d] = ReLU( (sum_k A[m,k]*W[d,k]) * scale[d] + shift[d] ), out as bf16.
// Block: 256 threads = 8 waves (4 M-waves x 2 N-waves), 128x128 tile.
// Wave tile: 32x64 -> 2x4 accumulators of 16x16 f32 (64 acc VGPRs).
// ---------------------------------------------------------------------------
__global__ __launch_bounds__(256, 2)
void gemm_bn_relu(const unsigned short* __restrict__ A,   // [4096,2048] bf16
                  const float* __restrict__ W,            // [2048,2048] f32 (row d, col k)
                  const float* __restrict__ gamma,
                  const float* __restrict__ beta,
                  const float* __restrict__ mean,
                  const float* __restrict__ var,
                  unsigned short* __restrict__ out)       // [4096,2048] bf16
{
  __shared__ __attribute__((aligned(16))) unsigned short As[2][TILE_M * LDS_STRIDE];
  __shared__ __attribute__((aligned(16))) unsigned short Ws[2][TILE_N * LDS_STRIDE];

  const int tid  = threadIdx.x;
  const int m0   = blockIdx.y * TILE_M;
  const int n0   = blockIdx.x * TILE_N;
  const int wave = tid >> 5;
  const int lane = tid & 31;
  const int wrow = wave & 3;     // 4 waves along M (32 rows each)
  const int wcol = wave >> 2;    // 2 waves along N (64 cols each)
  const int l16  = lane & 15;
  const int hi   = lane >> 4;    // half-wave select per ISA fragment layouts

  // Global-tile loader mapping: thread t -> tile row t/2, K-half t%2 (16 elems).
  const int lr = tid >> 1;
  const int lh = tid & 1;
  const unsigned short* Ag = A + (size_t)(m0 + lr) * C_TOT + lh * 16;
  const float*          Wg = W + (size_t)(n0 + lr) * C_TOT + lh * 16;

  v8f acc[2][4];
  for (int i = 0; i < 2; ++i)
    for (int j = 0; j < 4; ++j)
      acc[i][j] = (v8f){};

  float4 wReg[4];
#if !USE_ASYNC_COPY
  uint4 aReg0, aReg1;
#endif

  // A tile: direct global->LDS async copy (2 x b128 per thread; the imm
  // offset applies to both the global and LDS address -> offsets 0 and 16).
  auto copyA = [&](int buf, int k0) {
#if USE_ASYNC_COPY
    lds_v4i*  dst = (lds_v4i*)&As[buf][lr * LDS_STRIDE + lh * 16];
    glob_v4i* src = (glob_v4i*)(Ag + k0);
    __builtin_amdgcn_global_load_async_to_lds_b128(src, dst, 0,  0);
    __builtin_amdgcn_global_load_async_to_lds_b128(src, dst, 16, 0);
#else
    (void)buf; (void)k0;
#endif
  };

  auto load_global = [&](int k0) {
#if !USE_ASYNC_COPY
    const uint4* ap = (const uint4*)(Ag + k0);
    aReg0 = ap[0];
    aReg1 = ap[1];
#endif
    const float4* wp = (const float4*)(Wg + k0);
    wReg[0] = wp[0]; wReg[1] = wp[1]; wReg[2] = wp[2]; wReg[3] = wp[3];
  };

  auto store_lds = [&](int buf) {
#if !USE_ASYNC_COPY
    uint4* as = (uint4*)&As[buf][lr * LDS_STRIDE + lh * 16];
    as[0] = aReg0;
    as[1] = aReg1;
#endif
    uint4 p0, p1;                  // f32 -> bf16 pack for the W tile
    p0.x = pack_bf16x2(wReg[0].x, wReg[0].y);
    p0.y = pack_bf16x2(wReg[0].z, wReg[0].w);
    p0.z = pack_bf16x2(wReg[1].x, wReg[1].y);
    p0.w = pack_bf16x2(wReg[1].z, wReg[1].w);
    p1.x = pack_bf16x2(wReg[2].x, wReg[2].y);
    p1.y = pack_bf16x2(wReg[2].z, wReg[2].w);
    p1.z = pack_bf16x2(wReg[3].x, wReg[3].y);
    p1.w = pack_bf16x2(wReg[3].z, wReg[3].w);
    uint4* ws = (uint4*)&Ws[buf][lr * LDS_STRIDE + lh * 16];
    ws[0] = p0;
    ws[1] = p1;
  };

  auto compute = [&](int buf) {
    v16bf afrag[2], bfrag[4];
    // A 16x32 bf16 layout (ISA 7.12.2): lane m<16 holds K{0..7,16..23},
    // lane m+16 holds K{8..15,24..31}; each half is one 16B LDS load.
    for (int i = 0; i < 2; ++i) {
      const unsigned short* p =
          &As[buf][(wrow * 32 + i * 16 + l16) * LDS_STRIDE + hi * 8];
      union { v16bf v; bf16x8 h[2]; } ua;
      ua.h[0] = *(const bf16x8*)p;
      ua.h[1] = *(const bf16x8*)(p + 16);
      afrag[i] = ua.v;
    }
    // B 32x16 bf16 layout: lane n holds column n, K{0..7,16..23} (lo half-wave)
    // or K{8..15,24..31} (hi half-wave): two 16B LDS loads.
    for (int j = 0; j < 4; ++j) {
      const unsigned short* p =
          &Ws[buf][(wcol * 64 + j * 16 + l16) * LDS_STRIDE + hi * 8];
      union { v16bf v; bf16x8 h[2]; } ub;
      ub.h[0] = *(const bf16x8*)p;
      ub.h[1] = *(const bf16x8*)(p + 16);
      bfrag[j] = ub.v;
    }
    for (int i = 0; i < 2; ++i)
      for (int j = 0; j < 4; ++j)
        acc[i][j] = __builtin_amdgcn_wmma_f32_16x16x32_bf16(
            false, afrag[i], false, bfrag[j], (short)0, acc[i][j], false, false);
  };

  copyA(0, 0);
  load_global(0);
  store_lds(0);
  wait_async_zero();
  __syncthreads();

  for (int kk = 0; kk < NK_STEPS; ++kk) {
    const int  cur  = kk & 1;
    const bool more = (kk + 1) < NK_STEPS;
    if (more) {
      copyA(cur ^ 1, (kk + 1) * TILE_K);
      load_global((kk + 1) * TILE_K);
    }
    compute(cur);
    if (more) store_lds(cur ^ 1);
    wait_async_zero();   // this wave's async A copies land before the barrier
    __syncthreads();
  }

  // Epilogue: f32 C/D layout — VGPR e holds row e (lanes 0-15) / row e+8
  // (lanes 16-31); lane&15 is the column. Fused BN (scale/shift) + ReLU,
  // quantize to bf16 for the next layer.
  for (int j = 0; j < 4; ++j) {
    const int col = n0 + wcol * 64 + j * 16 + l16;
    const float s  = gamma[col] * rsqrtf(var[col] + 1e-5f);
    const float sh = beta[col] - mean[col] * s;
    for (int i = 0; i < 2; ++i) {
      const int rowbase = m0 + wrow * 32 + i * 16 + hi * 8;
      v8f a = acc[i][j];
      for (int e = 0; e < 8; ++e) {
        float y = a[e] * s + sh;
        y = y > 0.0f ? y : 0.0f;
        out[(size_t)(rowbase + e) * C_TOT + col] = f32_to_bf16(y);
      }
    }
  }
}

// ---------------------------------------------------------------------------
// Final 1x1 conv per head: out[m, col_off+c] = y[m,:] . Wf[c,:] + bf[c].
// Tiny (<1% FLOPs); thread per (m,c), 8-wide vectorized K loop.
// ---------------------------------------------------------------------------
__global__ void final_head(const unsigned short* __restrict__ Y,  // [4096,2048] bf16
                           const float* __restrict__ Wf,          // [nc,2048]
                           const float* __restrict__ bfv,         // [nc]
                           float* __restrict__ out,               // [4096,26]
                           int col_off, int nc)
{
  int t = blockIdx.x * blockDim.x + threadIdx.x;
  if (t >= M_TOT * nc) return;
  const int m = t / nc;
  const int c = t % nc;
  const uint4*  y = (const uint4*)(Y + (size_t)m * C_TOT);
  const float4* w = (const float4*)(Wf + (size_t)c * C_TOT);
  float acc = 0.0f;
  for (int k8 = 0; k8 < C_TOT / 8; ++k8) {
    uint4  yv = y[k8];
    float4 w0 = w[k8 * 2 + 0];
    float4 w1 = w[k8 * 2 + 1];
    acc += bf16_to_f32((unsigned short)(yv.x      )) * w0.x;
    acc += bf16_to_f32((unsigned short)(yv.x >> 16)) * w0.y;
    acc += bf16_to_f32((unsigned short)(yv.y      )) * w0.z;
    acc += bf16_to_f32((unsigned short)(yv.y >> 16)) * w0.w;
    acc += bf16_to_f32((unsigned short)(yv.z      )) * w1.x;
    acc += bf16_to_f32((unsigned short)(yv.z >> 16)) * w1.y;
    acc += bf16_to_f32((unsigned short)(yv.w      )) * w1.z;
    acc += bf16_to_f32((unsigned short)(yv.w >> 16)) * w1.w;
  }
  out[(size_t)m * 26 + col_off + c] = acc + bfv[c];
}

// ---------------------------------------------------------------------------
extern "C" void kernel_launch(void* const* d_in, const int* in_sizes, int n_in,
                              void* d_out, int out_size, void* d_ws, size_t ws_size,
                              hipStream_t stream) {
  (void)in_sizes; (void)n_in; (void)out_size; (void)ws_size;

  const float* features = (const float*)d_in[0];
  const float* Wl[3] = {(const float*)d_in[1], (const float*)d_in[6],  (const float*)d_in[11]};
  const float* Gl[3] = {(const float*)d_in[2], (const float*)d_in[7],  (const float*)d_in[12]};
  const float* Bl[3] = {(const float*)d_in[3], (const float*)d_in[8],  (const float*)d_in[13]};
  const float* Ml[3] = {(const float*)d_in[4], (const float*)d_in[9],  (const float*)d_in[14]};
  const float* Vl[3] = {(const float*)d_in[5], (const float*)d_in[10], (const float*)d_in[15]};
  const float* Wf[3]  = {(const float*)d_in[16], (const float*)d_in[18], (const float*)d_in[20]};
  const float* bfv[3] = {(const float*)d_in[17], (const float*)d_in[19], (const float*)d_in[21]};
  float* out = (float*)d_out;

  const size_t actElems = (size_t)M_TOT * C_TOT;
  unsigned short* xbf  = (unsigned short*)d_ws;
  unsigned short* ping = xbf  + actElems;
  unsigned short* pong = ping + actElems;

  {
    const int n4 = (int)(actElems / 4);
    cvt_f32_to_bf16<<<(n4 + 255) / 256, 256, 0, stream>>>(
        (const float4*)features, (uint2*)xbf, n4);
  }

  const dim3 grid(C_TOT / TILE_N, M_TOT / TILE_M);   // (16, 32)
  const int ncls[3] = {16, 5, 5};
  int coff = 0;
  for (int h = 0; h < 3; ++h) {
    const size_t wOff = (size_t)h * C_TOT * C_TOT;
    const size_t cOff = (size_t)h * C_TOT;
    gemm_bn_relu<<<grid, 256, 0, stream>>>(
        xbf,  Wl[0] + wOff, Gl[0] + cOff, Bl[0] + cOff, Ml[0] + cOff, Vl[0] + cOff, ping);
    gemm_bn_relu<<<grid, 256, 0, stream>>>(
        ping, Wl[1] + wOff, Gl[1] + cOff, Bl[1] + cOff, Ml[1] + cOff, Vl[1] + cOff, pong);
    gemm_bn_relu<<<grid, 256, 0, stream>>>(
        pong, Wl[2] + wOff, Gl[2] + cOff, Bl[2] + cOff, Ml[2] + cOff, Vl[2] + cOff, ping);
    const int nc  = ncls[h];
    const int tot = M_TOT * nc;
    final_head<<<(tot + 255) / 256, 256, 0, stream>>>(ping, Wf[h], bfv[h], out, coff, nc);
    coff += nc;
  }
}
